// EfficientKANConv2D_68032281969069
// MI455X (gfx1250) — matrix operand order repaired and checked by reading, not compiled, not run on hardware
//
#include <hip/hip_runtime.h>

// ---- types ----
typedef __attribute__((ext_vector_type(16))) __bf16 v16bf;
typedef __attribute__((ext_vector_type(8)))  __bf16 v8bf;
typedef __attribute__((ext_vector_type(8)))  float  v8f;

#define B_      8
#define CIN     64
#define HDIM    64
#define WDIM    64
#define COUT    64
#define INFEAT  576
#define KTOT    3456      // (1 silu + 5 spline) * 576
#define NFC     18        // feature chunks: INFEAT / 32
#define NG      6         // activation groups: silu + 5 spline bases
#define MTILE   128
#define ASTR    40        // padded LDS row pitch (halfs) for A tiles
#define WSTR    40        // padded LDS row pitch (halfs) for W tiles

// Pack combined weight matrix into bf16: Wb[n][k], k = blk*576 + f
// blk 0 -> base_weight[n][f]; blk j+1 -> spline_weight[n][f][j] * scaler[n][f]
__global__ __launch_bounds__(256)
void kan_prep_weights(const float* __restrict__ bw,
                      const float* __restrict__ sw,
                      const float* __restrict__ sc,
                      __bf16* __restrict__ Wb) {
  int idx = blockIdx.x * 256 + threadIdx.x;
  if (idx >= COUT * KTOT) return;
  int n   = idx / KTOT;
  int k   = idx - n * KTOT;
  int blk = k / INFEAT;
  int f   = k - blk * INFEAT;
  float v;
  if (blk == 0) v = bw[n * INFEAT + f];
  else          v = sw[(n * INFEAT + f) * 5 + (blk - 1)] * sc[n * INFEAT + f];
  Wb[idx] = (__bf16)v;
}

// Uniform quadratic B-spline segment (support [0,3), translates give the 5 bases).
__device__ __forceinline__ float quadb(float u) {
  float r = 0.0f;
  if (u >= 0.0f && u < 1.0f)        r = 0.5f * u * u;
  else if (u >= 1.0f && u < 2.0f) { float d = u - 1.5f; r = 0.75f - d * d; }
  else if (u >= 2.0f && u < 3.0f) { float d = 3.0f - u;  r = 0.5f * d * d; }
  return r;
}

// Fused im2col + KAN activation generation + bf16 WMMA GEMM over K=3456.
// Per feature chunk (32 features): load each x once, emit all 6 activation
// groups (silu + 5 spline bases) into LDS, then 6 WMMA sub-rounds.
// Grid: 256 blocks (M tiles of 128 rows), 256 threads (8 wave32).
__global__ __launch_bounds__(256)
void kan_conv_wmma(const float* __restrict__ x,
                   const __bf16* __restrict__ Wb,
                   float* __restrict__ out) {
  __shared__ __bf16 Alds[NG * MTILE * ASTR];  // 60 KB
  __shared__ __bf16 Wlds[NG * COUT * WSTR];   // 30 KB

  const int tid   = threadIdx.x;
  const int lane  = tid & 31;
  const int wave  = tid >> 5;
  const int l     = lane & 15;
  const int hi    = lane >> 4;            // 0: lanes 0-15, 1: lanes 16-31
  const int mbase = blockIdx.x * MTILE;

  v8f acc[4];
#pragma unroll
  for (int nt = 0; nt < 4; ++nt) acc[nt] = (v8f){0,0,0,0,0,0,0,0};

  const int t  = tid & 31;                // K-slot this thread stages
  const int r0 = tid >> 5;                // staging base row (0..7)
  const int wn   = tid >> 2;              // weight staging: row 0..63
  const int wseg = (tid & 3) * 8;         // 8-half segment within a 32-K slab

  for (int fc = 0; fc < NFC; ++fc) {
    const int fbase = fc * 32;

    // ---- stage 6 weight slabs: global bf16 -> LDS (b128 per thread/group) ----
#pragma unroll
    for (int g = 0; g < NG; ++g) {
      v8bf wv = *(const v8bf*)&Wb[wn * KTOT + g * INFEAT + fbase + wseg];
      *(v8bf*)&Wlds[(g * COUT + wn) * WSTR + wseg] = wv;
    }

    // ---- stage A: one x-load -> silu + 5 spline bases (6 LDS groups) ----
    const int f  = fbase + t;
    const int c  = f / 9;
    const int rf = f - c * 9;
    const int kh = rf / 3;
    const int kw = rf - kh * 3;
#pragma unroll
    for (int i = 0; i < 16; ++i) {
      const int row  = r0 + i * 8;
      const int nrow = mbase + row;
      const int b    = nrow >> 12;                 // HW = 4096
      const int hw   = nrow & 4095;
      const int h    = hw >> 6;
      const int w    = hw & 63;
      const int hs   = h + kh - 1;
      const int ws   = w + kw - 1;
      float xv = 0.0f;
      if (hs >= 0 && hs < HDIM && ws >= 0 && ws < WDIM)
        xv = x[((b * CIN + c) * HDIM + hs) * WDIM + ws];

      // silu
      float sig = 1.0f / (1.0f + __expf(-xv));
      Alds[row * ASTR + t] = (__bf16)(xv * sig);
      // 5 spline bases: u_j = u0 - j, u0 = (x - t_0)/h, t_0 = -2h-1, h = 2/3
      float u0 = (xv + 1.0f) * 1.5f + 2.0f;
#pragma unroll
      for (int j = 0; j < 5; ++j) {
        Alds[((j + 1) * MTILE + row) * ASTR + t] = (__bf16)quadb(u0 - (float)j);
      }
    }

    __syncthreads();

    // ---- WMMA: 6 K-blocks x 4 n-tiles per wave (16-row strip x 64 cols) ----
#pragma unroll
    for (int g = 0; g < NG; ++g) {
      v16bf afrag;
      {
        // 16-bit A 16x32 layout: lanes 0-15 K{0..7,16..23}; lanes 16-31 K{8..15,24..31}
        const __bf16* ap = &Alds[(g * MTILE + wave * 16 + l) * ASTR + hi * 8];
        v8bf alo = *(const v8bf*)ap;
        v8bf ahi = *(const v8bf*)(ap + 16);
#pragma unroll
        for (int q = 0; q < 8; ++q) { afrag[q] = alo[q]; afrag[q + 8] = ahi[q]; }
      }
#pragma unroll
      for (int nt = 0; nt < 4; ++nt) {
        // B 32x16: lane (l,hi) holds column N=l, K{hi*8..hi*8+7, 16+hi*8..16+hi*8+7}
        const __bf16* bp = &Wlds[(g * COUT + nt * 16 + l) * WSTR + hi * 8];
        v8bf blo = *(const v8bf*)bp;
        v8bf bhi = *(const v8bf*)(bp + 16);
        v16bf bfrag;
#pragma unroll
        for (int q = 0; q < 8; ++q) { bfrag[q] = blo[q]; bfrag[q + 8] = bhi[q]; }
        acc[nt] = __builtin_amdgcn_wmma_f32_16x16x32_bf16(
            false, afrag, false, bfrag, (short)0, acc[nt], false, false);
      }
    }

    __syncthreads();
  }

  // ---- epilogue: C/D layout -> out (B, C_OUT, H, W) ----
  const int rowbase = mbase + wave * 16;
#pragma unroll
  for (int nt = 0; nt < 4; ++nt) {
    const int ncol = nt * 16 + l;
#pragma unroll
    for (int i = 0; i < 8; ++i) {
      const int nrow = rowbase + hi * 8 + i;   // VGPR i: M = i (+8 for hi lanes)
      const int b    = nrow >> 12;
      const int hw   = nrow & 4095;
      out[(b * COUT + ncol) * 4096 + hw] = acc[nt][i];
    }
  }
}

extern "C" void kernel_launch(void* const* d_in, const int* in_sizes, int n_in,
                              void* d_out, int out_size, void* d_ws, size_t ws_size,
                              hipStream_t stream) {
  const float* x  = (const float*)d_in[0];   // (8,64,64,64)
  const float* bw = (const float*)d_in[1];   // (64,576)
  const float* sw = (const float*)d_in[2];   // (64,576,5)
  const float* sc = (const float*)d_in[3];   // (64,576)
  float* out = (float*)d_out;                // (8,64,64,64)
  __bf16* Wb = (__bf16*)d_ws;                // 64*3456 bf16 = 442 KB scratch

  kan_prep_weights<<<(COUT * KTOT + 255) / 256, 256, 0, stream>>>(bw, sw, sc, Wb);
  kan_conv_wmma<<<256, 256, 0, stream>>>(x, Wb, out);
}
